// GroupedQueryAttention_12438225289475
// MI455X (gfx1250) — compile-verified, matrix operand order checked by reading, not run on hardware
//
#include <hip/hip_runtime.h>

// ---------------------------------------------------------------------------
// Types matching the gfx1250 WMMA builtin signatures
// ---------------------------------------------------------------------------
typedef __bf16 bf16;
typedef __attribute__((ext_vector_type(16))) __bf16 v16bf;
typedef __attribute__((ext_vector_type(8)))  __bf16 v8bf;
typedef __attribute__((ext_vector_type(8)))  float  v8f;
typedef __attribute__((ext_vector_type(4)))  unsigned u32x4;
typedef __attribute__((ext_vector_type(8)))  unsigned u32x8;

#define WMMA_BF16(a, b, c) \
    __builtin_amdgcn_wmma_f32_16x16x32_bf16(false, (a), false, (b), (short)0, (c), false, false)

// ---------------------------------------------------------------------------
// 16-bit A-fragment loader (ISA 7.12.2, 16-bit A 16x32):
//   lanes 0-15 : row = lane,    K = {k0+0..7, k0+16..23}
//   lanes 16-31: row = lane-16, K = {k0+8..15, k0+24..31}
// B-fragment of 16x16x32 is the mirror (N<->M): loading B^T (row-major) with
// the same pattern gives the B operand.  Two 16-byte loads (global or ds).
// ---------------------------------------------------------------------------
static __device__ __forceinline__ v16bf load_frag(const bf16* base, int ld, int rc0, int k0, int lane)
{
    int r  = rc0 + (lane & 15);
    int kb = k0 + ((lane >> 4) << 3);
    const bf16* p = base + (size_t)r * (size_t)ld + kb;
    union { v16bf v; v8bf h[2]; } u;
    u.h[0] = *(const v8bf*)(p);
    u.h[1] = *(const v8bf*)(p + 16);
    return u.v;
}

// ---------------------------------------------------------------------------
// Tensor Data Mover: DMA a 2-D bf16 tile (tile_d1 rows x tile_d0 elems) from
// global memory into LDS.  Builds D# group0 (128b) + group1 (256b) per ISA
// cdna5 §8.3/§8.4 and issues the 2-operand form (VADDR2/3 = NULL -> 2D).
//   group0: [1:0]=count=1 | [63:32]=lds_addr | [120:64]=global_addr | type=2
//   group1: data_size=1 (2B), tensor_dim0/1, tile_dim0/1, tensor_dim0_stride
// Tracked by TENSORcnt (s_wait_tensorcnt).
// ---------------------------------------------------------------------------
static __device__ __forceinline__ void tdm_load_2d(unsigned lds_off, const bf16* gptr,
                                                   unsigned tile_d0, unsigned tile_d1,
                                                   unsigned tensor_d0, unsigned tensor_d1,
                                                   unsigned stride0 /* elements */)
{
    unsigned long long ga = (unsigned long long)gptr;
    u32x4 g0;
    g0[0] = 1u;                                        // count=1, user descriptor
    g0[1] = lds_off;                                   // lds_addr (bytes)
    g0[2] = (unsigned)ga;                              // global_addr[31:0]
    g0[3] = ((unsigned)(ga >> 32) & 0x01FFFFFFu) | (2u << 30);  // addr[56:32] | type=2

    u32x8 g1;
    g1[0] = 0x00010000u;                               // wg_mask=0, data_size=1 (2B)
    g1[1] = (tensor_d0 & 0xFFFFu) << 16;               // tensor_dim0 bits[79:48]
    g1[2] = (tensor_d0 >> 16) | ((tensor_d1 & 0xFFFFu) << 16);   // ..| tensor_dim1[111:80]
    g1[3] = (tensor_d1 >> 16) | (tile_d0 << 16);       // ..| tile_dim0[127:112]
    g1[4] = tile_d1;                                   // tile_dim1[143:128], tile_dim2=0
    g1[5] = stride0;                                   // tensor_dim0_stride[191:160]
    g1[6] = 0u;                                        // stride hi / dim1_stride lo
    g1[7] = 0u;                                        // dim1_stride hi

    asm volatile("tensor_load_to_lds %0, %1" :: "s"(g0), "s"(g1) : "memory");
}

// ---------------------------------------------------------------------------
// 64x64 wave tile GEMM core: acc[4][4] of v8f, K-loop of 16x16x32 WMMAs.
// A: row-major MxK, Bt: row-major NxK (i.e. B transposed = W row-major).
// ---------------------------------------------------------------------------
static __device__ __forceinline__ void gemm_tile_64x64(const bf16* __restrict__ A,
                                                       const bf16* __restrict__ Bt,
                                                       int K, int row0, int col0, int lane,
                                                       v8f acc[4][4])
{
#pragma unroll 1
    for (int k0 = 0; k0 < K; k0 += 32) {
        v16bf af[4], bw[4];
#pragma unroll
        for (int mi = 0; mi < 4; ++mi) af[mi] = load_frag(A,  K, row0 + mi * 16, k0, lane);
#pragma unroll
        for (int ni = 0; ni < 4; ++ni) bw[ni] = load_frag(Bt, K, col0 + ni * 16, k0, lane);
        if (k0 + 32 < K) {  // global_prefetch_b8 of the next K-slab
            __builtin_prefetch(A  + (size_t)(row0 + (lane & 15)) * K + k0 + 32, 0, 3);
            __builtin_prefetch(Bt + (size_t)(col0 + (lane & 15)) * K + k0 + 32, 0, 3);
        }
#pragma unroll
        for (int mi = 0; mi < 4; ++mi)
#pragma unroll
            for (int ni = 0; ni < 4; ++ni)
                acc[mi][ni] = WMMA_BF16(af[mi], bw[ni], acc[mi][ni]);
    }
}

// ---------------------------------------------------------------------------
// f32 -> bf16 conversion (grid-stride)
// ---------------------------------------------------------------------------
__global__ void convert_f32_to_bf16(const float* __restrict__ src, bf16* __restrict__ dst, int n)
{
    int i = blockIdx.x * blockDim.x + threadIdx.x;
    int stride = gridDim.x * blockDim.x;
    for (; i < n; i += stride) dst[i] = (bf16)src[i];
}

// ---------------------------------------------------------------------------
// Fused QKV projection + RoPE.
//   X  : [4096, 2048] bf16 (B*T rows)
//   Wf : [3072, 2048] bf16 = rows [Wq(2048); Wk(512); Wv(512)]  (B^T layout)
//   Qb : [B,H,T,D]  bf16 (RoPE applied)
//   Kb : [B,HKV,T,D] bf16 (RoPE applied)
//   Vt : [B,HKV,D,T] bf16 (transposed so attn@V B-fragments are row-major)
// Block = 256 threads = 8 waves as 2(M)x4(N); block tile 128x256; wave 64x64.
// Wave N-tile (64) is head-aligned, so RoPE pair (c, c+32) = acc tiles j, j+2
// in the SAME lane, same angle (inv_freq[(c+32)%32] == inv_freq[c]).
// ---------------------------------------------------------------------------
__global__ __launch_bounds__(256) void qkv_rope_kernel(const bf16* __restrict__ X,
                                                       const bf16* __restrict__ Wf,
                                                       bf16* __restrict__ Qb,
                                                       bf16* __restrict__ Kb,
                                                       bf16* __restrict__ Vt)
{
    constexpr int T = 2048, D = 64, H = 32, HKV = 8, K = 2048;
    constexpr float L2_BASE_OVER_HALFD = 0.41524101186f;  // log2(10000)/32

    int lane = threadIdx.x & 31;
    int w    = threadIdx.x >> 5;
    int row0 = blockIdx.x * 128 + (w >> 2) * 64;
    int col0 = blockIdx.y * 256 + (w & 3) * 64;

    v8f acc[4][4] = {};
    gemm_tile_64x64(X, Wf, K, row0, col0, lane, acc);

    int ln = lane & 15, hh = lane >> 4;

    if (col0 < 2048) {                       // ---- Q region, RoPE ----
        int h = col0 >> 6;
        float inv0 = exp2f(-(float)(ln)      * L2_BASE_OVER_HALFD);
        float inv1 = exp2f(-(float)(16 + ln) * L2_BASE_OVER_HALFD);
#pragma unroll
        for (int mi = 0; mi < 4; ++mi)
#pragma unroll
            for (int r = 0; r < 8; ++r) {
                int m = row0 + mi * 16 + r + 8 * hh;
                int b = m >> 11, t = m & 2047;
                bf16* dst = Qb + ((size_t)(b * H + h) * T + t) * D;
                float s0, c0, s1, c1;
                __sincosf((float)t * inv0, &s0, &c0);
                __sincosf((float)t * inv1, &s1, &c1);
                float qa0 = acc[mi][0][r], qb0 = acc[mi][2][r];
                float qa1 = acc[mi][1][r], qb1 = acc[mi][3][r];
                dst[ln]      = (bf16)(qa0 * c0 - qb0 * s0);
                dst[16 + ln] = (bf16)(qa1 * c1 - qb1 * s1);
                dst[32 + ln] = (bf16)(qb0 * c0 + qa0 * s0);
                dst[48 + ln] = (bf16)(qb1 * c1 + qa1 * s1);
            }
    } else if (col0 < 2560) {                // ---- K region, RoPE ----
        int hk = (col0 - 2048) >> 6;
        float inv0 = exp2f(-(float)(ln)      * L2_BASE_OVER_HALFD);
        float inv1 = exp2f(-(float)(16 + ln) * L2_BASE_OVER_HALFD);
#pragma unroll
        for (int mi = 0; mi < 4; ++mi)
#pragma unroll
            for (int r = 0; r < 8; ++r) {
                int m = row0 + mi * 16 + r + 8 * hh;
                int b = m >> 11, t = m & 2047;
                bf16* dst = Kb + ((size_t)(b * HKV + hk) * T + t) * D;
                float s0, c0, s1, c1;
                __sincosf((float)t * inv0, &s0, &c0);
                __sincosf((float)t * inv1, &s1, &c1);
                float ka0 = acc[mi][0][r], kb0 = acc[mi][2][r];
                float ka1 = acc[mi][1][r], kb1 = acc[mi][3][r];
                dst[ln]      = (bf16)(ka0 * c0 - kb0 * s0);
                dst[16 + ln] = (bf16)(ka1 * c1 - kb1 * s1);
                dst[32 + ln] = (bf16)(kb0 * c0 + ka0 * s0);
                dst[48 + ln] = (bf16)(kb1 * c1 + ka1 * s1);
            }
    } else {                                 // ---- V region, store transposed ----
        int hk = (col0 - 2560) >> 6;
#pragma unroll
        for (int mi = 0; mi < 4; ++mi)
#pragma unroll
            for (int r = 0; r < 8; ++r) {
                int m = row0 + mi * 16 + r + 8 * hh;
                int b = m >> 11, t = m & 2047;
#pragma unroll
                for (int j = 0; j < 4; ++j) {
                    int d = j * 16 + ln;
                    Vt[((size_t)(b * HKV + hk) * D + d) * T + t] = (bf16)acc[mi][j][r];
                }
            }
    }
}

// ---------------------------------------------------------------------------
// Causal flash attention with GQA (kv head = h/4).
//   Qb [B,H,T,D], Kb [B,HKV,T,D], Vt [B,HKV,D,T]  -> Ob [B*T, H*D] bf16
// Block = 256 = 8 waves; wave owns 16 q-rows; key tiles of 64.
// Per key tile the TDM DMAs the 64x64 K-tile and V-tile into this wave's LDS
// staging buffers (tensor_load_to_lds, TENSORcnt); WMMA fragments then come
// from ds_load_b128.  Row statistics reduced with shfl_xor over the 16-lane
// groups of the C-matrix layout; P converts C-layout -> A-layout via LDS.
// ---------------------------------------------------------------------------
__global__ __launch_bounds__(256) void flash_attn_kernel(const bf16* __restrict__ Qb,
                                                         const bf16* __restrict__ Kb,
                                                         const bf16* __restrict__ Vt,
                                                         bf16* __restrict__ Ob)
{
    constexpr int T = 2048, D = 64, H = 32, HKV = 8;
    constexpr int LDP = 72;                    // padded row (16B aligned rows)
    __shared__ bf16 Ksh[8][64][64];            // per-wave K tile  (keys x d)
    __shared__ bf16 Vsh[8][64][64];            // per-wave V tile  (d x keys)
    __shared__ bf16 P[8][16][LDP];             // per-wave P tile

    int lane = threadIdx.x & 31;
    int w    = threadIdx.x >> 5;
    int ln = lane & 15, hh = lane >> 4;
    int q0 = blockIdx.x * 128 + w * 16;
    int h = blockIdx.y, b = blockIdx.z, hk = h >> 2;

    const bf16* Qh = Qb + (size_t)(b * H   + h ) * T * D;
    const bf16* Kh = Kb + (size_t)(b * HKV + hk) * T * D;
    const bf16* Vh = Vt + (size_t)(b * HKV + hk) * D * T;

    const bf16* Kw = &Ksh[w][0][0];
    const bf16* Vw = &Vsh[w][0][0];
    // generic LDS address: addr[31:0] is the LDS byte offset (ISA 10.2)
    unsigned offK = (unsigned)(size_t)(const void*)Kw;
    unsigned offV = (unsigned)(size_t)(const void*)Vw;

    v16bf aq0 = load_frag(Qh, D, q0, 0,  lane);
    v16bf aq1 = load_frag(Qh, D, q0, 32, lane);

    v8f o[4] = {};
    float mrow[8], lrow[8];
#pragma unroll
    for (int r = 0; r < 8; ++r) { mrow[r] = -1e30f; lrow[r] = 0.f; }

    int ktmax = (q0 + 16 + 63) >> 6;
#pragma unroll 1
    for (int kt = 0; kt < ktmax; ++kt) {
        // drain outstanding ds reads of the staging buffers, then DMA new tiles
        asm volatile("s_wait_dscnt 0x0" ::: "memory");
        tdm_load_2d(offK, Kh + (size_t)kt * 64 * D, /*tile*/ 64, 64,
                    /*tensor*/ (unsigned)D, (unsigned)T, /*stride0*/ (unsigned)D);
        tdm_load_2d(offV, Vh + (size_t)kt * 64,     /*tile*/ 64, 64,
                    /*tensor*/ (unsigned)T, (unsigned)D, /*stride0*/ (unsigned)T);
        __builtin_amdgcn_s_wait_tensorcnt(0);

        // S = Q @ K^T  (16 x 64 tile) from the staged K tile
        v8f s[4] = {};
#pragma unroll
        for (int j = 0; j < 4; ++j)
            s[j] = WMMA_BF16(aq0, load_frag(Kw, 64, j * 16, 0,  lane), s[j]);
#pragma unroll
        for (int j = 0; j < 4; ++j)
            s[j] = WMMA_BF16(aq1, load_frag(Kw, 64, j * 16, 32, lane), s[j]);

        // scale, causal mask, online softmax
#pragma unroll
        for (int r = 0; r < 8; ++r) {
            int mg = q0 + r + 8 * hh;
            float rowm = -1e30f;
#pragma unroll
            for (int j = 0; j < 4; ++j) {
                int ng = kt * 64 + j * 16 + ln;
                float v = s[j][r] * 0.125f;            // 1/sqrt(64)
                v = (ng <= mg) ? v : -1e30f;
                s[j][r] = v;
                rowm = fmaxf(rowm, v);
            }
            rowm = fmaxf(rowm, __shfl_xor(rowm, 1, 32));
            rowm = fmaxf(rowm, __shfl_xor(rowm, 2, 32));
            rowm = fmaxf(rowm, __shfl_xor(rowm, 4, 32));
            rowm = fmaxf(rowm, __shfl_xor(rowm, 8, 32));
            float mnew  = fmaxf(mrow[r], rowm);
            float alpha = __expf(mrow[r] - mnew);
            float ps = 0.f;
#pragma unroll
            for (int j = 0; j < 4; ++j) {
                float p = __expf(s[j][r] - mnew);
                s[j][r] = p;
                ps += p;
            }
            ps += __shfl_xor(ps, 1, 32);
            ps += __shfl_xor(ps, 2, 32);
            ps += __shfl_xor(ps, 4, 32);
            ps += __shfl_xor(ps, 8, 32);
            lrow[r] = lrow[r] * alpha + ps;
            mrow[r] = mnew;
#pragma unroll
            for (int j = 0; j < 4; ++j) o[j][r] *= alpha;

            // spill P (C-layout) to this wave's LDS tile, row-major 16x64
            int m = r + 8 * hh;
#pragma unroll
            for (int j = 0; j < 4; ++j) P[w][m][j * 16 + ln] = (bf16)s[j][r];
        }

        // reload P as A-fragments and accumulate O += P @ V (staged V tile)
        const bf16* Pp = &P[w][0][0];
        v16bf ap0 = load_frag(Pp, LDP, 0, 0,  lane);
        v16bf ap1 = load_frag(Pp, LDP, 0, 32, lane);
#pragma unroll
        for (int j = 0; j < 4; ++j)
            o[j] = WMMA_BF16(ap0, load_frag(Vw, 64, j * 16, 0,  lane), o[j]);
#pragma unroll
        for (int j = 0; j < 4; ++j)
            o[j] = WMMA_BF16(ap1, load_frag(Vw, 64, j * 16, 32, lane), o[j]);
    }

    // normalize and store to [b*T+t, h*64+d] for the output projection
#pragma unroll
    for (int r = 0; r < 8; ++r) {
        float li = 1.0f / lrow[r];
        int t = q0 + r + 8 * hh;
        bf16* dst = Ob + (size_t)(b * T + t) * 2048 + h * 64;
#pragma unroll
        for (int j = 0; j < 4; ++j) dst[j * 16 + ln] = (bf16)(o[j][r] * li);
    }
}

// ---------------------------------------------------------------------------
// Output projection: out = Ob @ Wo^T   (M=4096, N=2048, K=2048), f32 out.
// ---------------------------------------------------------------------------
__global__ __launch_bounds__(256) void out_proj_kernel(const bf16* __restrict__ Ob,
                                                       const bf16* __restrict__ Wob,
                                                       float* __restrict__ out)
{
    constexpr int K = 2048, N = 2048;
    int lane = threadIdx.x & 31;
    int w    = threadIdx.x >> 5;
    int row0 = blockIdx.x * 128 + (w >> 2) * 64;
    int col0 = blockIdx.y * 256 + (w & 3) * 64;

    v8f acc[4][4] = {};
    gemm_tile_64x64(Ob, Wob, K, row0, col0, lane, acc);

    int ln = lane & 15, hh = lane >> 4;
#pragma unroll
    for (int mi = 0; mi < 4; ++mi)
#pragma unroll
        for (int r = 0; r < 8; ++r) {
            int m = row0 + mi * 16 + r + 8 * hh;
#pragma unroll
            for (int ni = 0; ni < 4; ++ni) {
                int n = col0 + ni * 16 + ln;
                out[(size_t)m * N + n] = acc[mi][ni][r];
            }
        }
}

// ---------------------------------------------------------------------------
// Launcher
// ---------------------------------------------------------------------------
extern "C" void kernel_launch(void* const* d_in, const int* in_sizes, int n_in,
                              void* d_out, int out_size, void* d_ws, size_t ws_size,
                              hipStream_t stream)
{
    const float* x  = (const float*)d_in[0];   // [2,2048,2048]
    const float* Wq = (const float*)d_in[1];   // [2048,2048]
    const float* Wk = (const float*)d_in[2];   // [512,2048]
    const float* Wv = (const float*)d_in[3];   // [512,2048]
    const float* Wo = (const float*)d_in[4];   // [2048,2048]

    char*  ws  = (char*)d_ws;
    size_t off = 0;
    auto take = [&](size_t bytes) -> void* {
        off = (off + 255) & ~(size_t)255;
        void* p = ws + off;
        off += bytes;
        return p;
    };

    bf16* xb  = (bf16*)take((size_t)4096 * 2048 * 2);          // x bf16
    bf16* Wf  = (bf16*)take((size_t)3072 * 2048 * 2);          // fused Wq|Wk|Wv bf16
    bf16* Wob = (bf16*)take((size_t)2048 * 2048 * 2);          // Wo bf16
    bf16* Qb  = (bf16*)take((size_t)2 * 32 * 2048 * 64 * 2);   // Q (RoPE'd)
    bf16* Kb  = (bf16*)take((size_t)2 * 8 * 2048 * 64 * 2);    // K (RoPE'd)
    bf16* Vtb = (bf16*)take((size_t)2 * 8 * 64 * 2048 * 2);    // V transposed
    bf16* Ob  = (bf16*)take((size_t)4096 * 2048 * 2);          // attention output

    convert_f32_to_bf16<<<512, 256, 0, stream>>>(x,  xb,  4096 * 2048);
    convert_f32_to_bf16<<<512, 256, 0, stream>>>(Wq, Wf,                       2048 * 2048);
    convert_f32_to_bf16<<<128, 256, 0, stream>>>(Wk, Wf + (size_t)2048 * 2048,  512 * 2048);
    convert_f32_to_bf16<<<128, 256, 0, stream>>>(Wv, Wf + (size_t)2560 * 2048,  512 * 2048);
    convert_f32_to_bf16<<<512, 256, 0, stream>>>(Wo, Wob,                      2048 * 2048);

    qkv_rope_kernel  <<<dim3(32, 12),    256, 0, stream>>>(xb, Wf, Qb, Kb, Vtb);
    flash_attn_kernel<<<dim3(16, 32, 2), 256, 0, stream>>>(Qb, Kb, Vtb, Ob);
    out_proj_kernel  <<<dim3(32, 8),     256, 0, stream>>>(Ob, Wob, (float*)d_out);
}
